// GraphAttentionLayer_28106265985312
// MI455X (gfx1250) — matrix operand order repaired and checked by reading, not compiled, not run on hardware
//
#include <hip/hip_runtime.h>
#include <hip/hip_bf16.h>
#include <math.h>

#define N_NODES 50000
#define N_EDGES 800000
#define IN_F 128
#define OUT_F 128
#define HEADS 8
#define HEAD_DIM 16
#define EDGE_F 32

typedef __attribute__((ext_vector_type(16))) __bf16 v16bf;
typedef __attribute__((ext_vector_type(8)))  float  v8f;

// TDM / async path availability (this toolchain: 6-arg tensor builtin)
#if defined(__has_builtin)
#if __has_builtin(__builtin_amdgcn_tensor_load_to_lds) && \
    __has_builtin(__builtin_amdgcn_s_wait_tensorcnt)
#define USE_TDM 1
#endif
#endif
#ifndef USE_TDM
#define USE_TDM 0
#endif

typedef unsigned int u32x4 __attribute__((ext_vector_type(4)));
typedef int          i32x4 __attribute__((ext_vector_type(4)));
typedef int          i32x8 __attribute__((ext_vector_type(8)));

#define ROWS_PER_BLOCK 64
#define LDS_STRIDE 132   // f32 row stride in LDS: 128 data + 4 pad DWORDs

// ---------------------------------------------------------------------------
// GEMM: out[M x 128] = A[M x 128] @ W[128 x 128] + bias, via bf16 WMMA.
// 64 rows per block, 8 waves; each wave owns a 16-col stripe and 4 row-tiles,
// reusing each B fragment across 4 WMMAs (16 WMMAs/wave).
// A tile staged to LDS as f32 by the Tensor Data Mover (padded via D#
// pad fields) when available, else by cooperative loads.
// ---------------------------------------------------------------------------
__global__ __launch_bounds__(256)
void gemm_bias_wmma(const float* __restrict__ A, const float* __restrict__ W,
                    const float* __restrict__ bias, float* __restrict__ out,
                    int M) {
  __shared__ float sA[ROWS_PER_BLOCK * LDS_STRIDE];
  const int row0 = blockIdx.x * ROWS_PER_BLOCK;
  const int t = threadIdx.x;

#if USE_TDM
  if (t < 32) {  // one wave issues the tensor DMA (EXEC ignored by TDM)
    const unsigned lds_addr = (unsigned)(unsigned long long)(void*)sA;
    const unsigned long long gaddr =
        (unsigned long long)(const void*)(A + (size_t)row0 * IN_F);
    const unsigned rows_left = (unsigned)(M - row0);  // OOB rows -> zero-fill

    u32x4 g0;
    g0[0] = 1u;                                   // count=1, load, no gather
    g0[1] = lds_addr;                             // lds_addr[31:0]
    g0[2] = (unsigned)(gaddr & 0xffffffffull);    // global_addr[31:0]
    g0[3] = (unsigned)((gaddr >> 32) & 0x01ffffffull) | (2u << 30);  // type=2

    i32x8 g1;
    g1[0] = (int)((2u << 16)      // data_size = 4B
                | (1u << 20)      // pad_enable
                | (6u << 22)      // pad_interval: 128 DWORDs
                | (3u << 25));    // pad_amount: 4 DWORDs  -> stride 132
    g1[1] = (int)((unsigned)IN_F << 16);          // tensor_dim0 = 128
    g1[2] = (int)((rows_left & 0xffffu) << 16);   // tensor_dim1 lo16
    g1[3] = (int)(((rows_left >> 16) & 0xffffu) | ((unsigned)IN_F << 16));
                                                  // tensor_dim1 hi16, tile_dim0=128
    g1[4] = ROWS_PER_BLOCK;                       // tile_dim1 = 64, tile_dim2 = 0
    g1[5] = IN_F;                                 // tensor_dim0_stride = 128
    g1[6] = 0;
    g1[7] = 0;

    i32x4 g2 = {0, 0, 0, 0};
    i32x4 g3 = {0, 0, 0, 0};
    i32x8 g4 = {0, 0, 0, 0, 0, 0, 0, 0};
    __builtin_amdgcn_tensor_load_to_lds(g0, g1, g2, g3, g4, 0);
    __builtin_amdgcn_s_wait_tensorcnt(0);
  }
  __syncthreads();
#else
  {  // cooperative staging: thread t -> row t>>2, 32 cols
    const int r  = t >> 2;
    const int c0 = (t & 3) * 32;
    const int gr = min(row0 + r, M - 1);
    const float4* src = (const float4*)(A + (size_t)gr * IN_F + c0);
    float4* dst = (float4*)&sA[r * LDS_STRIDE + c0];
#pragma unroll
    for (int j = 0; j < 8; ++j) dst[j] = src[j];
  }
  __syncthreads();
#endif

  const int wave = t >> 5;            // 0..7 -> output column stripe
  const int lane = t & 31;
  const int m    = lane & 15;         // row within 16-row tile
  const int kh   = lane >> 4;         // K-half selector (this code's contiguous k order)
  const int col  = wave * 16 + (lane & 15);

  v8f acc[4];
  const float bcol = bias[col];
#pragma unroll
  for (int ti = 0; ti < 4; ++ti)
#pragma unroll
    for (int r = 0; r < 8; ++r) acc[ti][r] = bcol;

#pragma unroll
  for (int ks = 0; ks < 4; ++ks) {
    const int k0 = ks * 32;
    // B fragment (32x16 bf16): elem e -> K=k0+16*kh+e, column = col
    v16bf b;
    const float* wcol = W + (size_t)(k0 + 16 * kh) * OUT_F + col;
#pragma unroll
    for (int e = 0; e < 16; ++e) b[e] = (__bf16)wcol[(size_t)e * OUT_F];
    if (ks < 3)
      __builtin_prefetch(W + (size_t)(k0 + 32) * OUT_F + col, 0, 1);
    // reuse B across the 4 row-tiles
#pragma unroll
    for (int ti = 0; ti < 4; ++ti) {
      // A fragment (16x32 bf16): elem j -> K=k0+16*kh+j
      v16bf a;
      const float* ar = &sA[(ti * 16 + m) * LDS_STRIDE + k0 + 16 * kh];
#pragma unroll
      for (int j = 0; j < 8; ++j) {
        a[j]     = (__bf16)ar[j];
        a[8 + j] = (__bf16)ar[8 + j];
      }
      acc[ti] = __builtin_amdgcn_wmma_f32_16x16x32_bf16(
          /*neg_a=*/false, a, /*neg_b=*/false, b,
          /*c_mod=*/(short)0, acc[ti], /*reuse_a=*/false, /*reuse_b=*/false);
    }
  }

  // D layout: VGPR r -> row (r + 8*kh) within tile
#pragma unroll
  for (int ti = 0; ti < 4; ++ti)
#pragma unroll
    for (int r = 0; r < 8; ++r) {
      const int row = row0 + ti * 16 + r + 8 * kh;
      if (row < M) out[(size_t)row * OUT_F + col] = acc[ti][r];
    }
}

// ---------------------------------------------------------------------------
// Edge bias: scores[e,h] = edge_features[e,:] @ We[:,h] + be[h]
// ---------------------------------------------------------------------------
__global__ __launch_bounds__(256)
void edge_bias_kernel(const float* __restrict__ ef, const float* __restrict__ We,
                      const float* __restrict__ be, float* __restrict__ scores) {
  const int e = blockIdx.x * blockDim.x + threadIdx.x;
  if (e >= N_EDGES) return;
  float x[EDGE_F];
  const float4* p = (const float4*)(ef + (size_t)e * EDGE_F);
#pragma unroll
  for (int j = 0; j < EDGE_F / 4; ++j) {
    float4 v = p[j];
    x[4 * j + 0] = v.x; x[4 * j + 1] = v.y; x[4 * j + 2] = v.z; x[4 * j + 3] = v.w;
  }
#pragma unroll
  for (int h = 0; h < HEADS; ++h) {
    float acc = be[h];
#pragma unroll
    for (int k = 0; k < EDGE_F; ++k) acc = fmaf(x[k], We[k * HEADS + h], acc);
    scores[(size_t)e * HEADS + h] = acc;
  }
}

// monotone float<->uint key for atomic max
__device__ __forceinline__ unsigned fkey(float f) {
  unsigned u = __float_as_uint(f);
  return (u & 0x80000000u) ? ~u : (u | 0x80000000u);
}
__device__ __forceinline__ float funkey(unsigned k) {
  return (k & 0x80000000u) ? __uint_as_float(k & 0x7fffffffu) : __uint_as_float(~k);
}

// ---------------------------------------------------------------------------
// Per (edge, head): score = dot(Q[t,h,:], K[s,h,:]) * scale + edge_bias,
// then atomic segment-max over target node.
// ---------------------------------------------------------------------------
__global__ __launch_bounds__(256)
void edge_score_kernel(const float* __restrict__ Q, const float* __restrict__ K,
                       const int* __restrict__ src, const int* __restrict__ dst,
                       float* __restrict__ scores, unsigned* __restrict__ maxkey) {
  const int i = blockIdx.x * blockDim.x + threadIdx.x;
  if (i >= N_EDGES * HEADS) return;
  const int e = i >> 3, h = i & 7;
  const int s = src[e], tg = dst[e];
  const float4* q = (const float4*)(Q + (size_t)tg * OUT_F + h * HEAD_DIM);
  const float4* k = (const float4*)(K + (size_t)s  * OUT_F + h * HEAD_DIM);
  float acc = 0.0f;
#pragma unroll
  for (int j = 0; j < HEAD_DIM / 4; ++j) {
    float4 qa = q[j], ka = k[j];
    acc = fmaf(qa.x, ka.x, acc); acc = fmaf(qa.y, ka.y, acc);
    acc = fmaf(qa.z, ka.z, acc); acc = fmaf(qa.w, ka.w, acc);
  }
  const float sc = fmaf(acc, 0.25f /* HEAD_DIM^-0.5 */, scores[i]);
  scores[i] = sc;
  atomicMax(&maxkey[(size_t)tg * HEADS + h], fkey(sc));
}

// ---------------------------------------------------------------------------
// Per (edge, head): p = exp(score - segmax); atomic segment-sum of p.
// ---------------------------------------------------------------------------
__global__ __launch_bounds__(256)
void edge_prob_kernel(const int* __restrict__ dst, const unsigned* __restrict__ maxkey,
                      float* __restrict__ scores, float* __restrict__ denom) {
  const int i = blockIdx.x * blockDim.x + threadIdx.x;
  if (i >= N_EDGES * HEADS) return;
  const int e = i >> 3, h = i & 7;
  const int tg = dst[e];
  const float mx = funkey(maxkey[(size_t)tg * HEADS + h]);
  const float p = __expf(scores[i] - mx);
  scores[i] = p;
  atomicAdd(&denom[(size_t)tg * HEADS + h], p);
}

// ---------------------------------------------------------------------------
// Per (edge, feature): agg[t,d] += (p/denom) * V[s,d]
// ---------------------------------------------------------------------------
__global__ __launch_bounds__(256)
void edge_aggregate_kernel(const int* __restrict__ src, const int* __restrict__ dst,
                           const float* __restrict__ probs, const float* __restrict__ denom,
                           const float* __restrict__ V, float* __restrict__ agg) {
  const long long i = (long long)blockIdx.x * blockDim.x + threadIdx.x;
  if (i >= (long long)N_EDGES * OUT_F) return;
  const int e = (int)(i >> 7);
  const int d = (int)(i & 127);
  const int h = d >> 4;
  const int s = src[e], tg = dst[e];
  const float w = probs[(size_t)e * HEADS + h] /
                  (denom[(size_t)tg * HEADS + h] + 1e-10f);
  atomicAdd(&agg[(size_t)tg * OUT_F + d], w * V[(size_t)s * OUT_F + d]);
}

// ---------------------------------------------------------------------------
// Workspace layout (float units)
// ---------------------------------------------------------------------------
static constexpr size_t OFF_Q    = 0;
static constexpr size_t OFF_K    = OFF_Q   + (size_t)N_NODES * OUT_F;
static constexpr size_t OFF_V    = OFF_K   + (size_t)N_NODES * OUT_F;
static constexpr size_t OFF_SC   = OFF_V   + (size_t)N_NODES * OUT_F;   // N_EDGES*HEADS
static constexpr size_t OFF_AGG  = OFF_SC  + (size_t)N_EDGES * HEADS;   // zeroed
static constexpr size_t OFF_MAXK = OFF_AGG + (size_t)N_NODES * OUT_F;   // zeroed
static constexpr size_t OFF_DEN  = OFF_MAXK + (size_t)N_NODES * HEADS;  // zeroed
static constexpr size_t ZERO_FLOATS =
    (size_t)N_NODES * OUT_F + 2 * (size_t)N_NODES * HEADS;

extern "C" void kernel_launch(void* const* d_in, const int* in_sizes, int n_in,
                              void* d_out, int out_size, void* d_ws, size_t ws_size,
                              hipStream_t stream) {
  const float* node_features = (const float*)d_in[0];
  const int*   edge_index    = (const int*)  d_in[1];   // [2, N_EDGES]
  const float* edge_features = (const float*)d_in[2];
  const float* Wq = (const float*)d_in[3];  const float* bq = (const float*)d_in[4];
  const float* Wk = (const float*)d_in[5];  const float* bk = (const float*)d_in[6];
  const float* Wv = (const float*)d_in[7];  const float* bv = (const float*)d_in[8];
  const float* We = (const float*)d_in[9];  const float* be = (const float*)d_in[10];
  const float* Wo = (const float*)d_in[11]; const float* bo = (const float*)d_in[12];

  float*    ws    = (float*)d_ws;
  float*    Q     = ws + OFF_Q;
  float*    Kp    = ws + OFF_K;
  float*    V     = ws + OFF_V;
  float*    SC    = ws + OFF_SC;
  float*    AGG   = ws + OFF_AGG;
  unsigned* MAXK  = (unsigned*)(ws + OFF_MAXK);
  float*    DEN   = ws + OFF_DEN;
  const int* srcs = edge_index;
  const int* dsts = edge_index + N_EDGES;

  // zero agg / segment-max keys / denom (contiguous region)
  (void)hipMemsetAsync(AGG, 0, ZERO_FLOATS * sizeof(float), stream);

  const dim3 blk(256);
  const dim3 gGemm((N_NODES + ROWS_PER_BLOCK - 1) / ROWS_PER_BLOCK);  // 782
  const dim3 gEdge(N_EDGES / 256);                                    // 3125
  const dim3 gEH((N_EDGES * HEADS) / 256);                            // 25000
  const dim3 gED((unsigned)(((long long)N_EDGES * OUT_F) / 256));     // 400000

  // dense projections (bf16 WMMA, f32 accumulate)
  gemm_bias_wmma<<<gGemm, blk, 0, stream>>>(node_features, Wq, bq, Q, N_NODES);
  gemm_bias_wmma<<<gGemm, blk, 0, stream>>>(node_features, Wk, bk, Kp, N_NODES);
  gemm_bias_wmma<<<gGemm, blk, 0, stream>>>(node_features, Wv, bv, V, N_NODES);

  // edge attention
  edge_bias_kernel<<<gEdge, blk, 0, stream>>>(edge_features, We, be, SC);
  edge_score_kernel<<<gEH, blk, 0, stream>>>(Q, Kp, srcs, dsts, SC, MAXK);
  edge_prob_kernel<<<gEH, blk, 0, stream>>>(dsts, MAXK, SC, DEN);
  edge_aggregate_kernel<<<gED, blk, 0, stream>>>(srcs, dsts, SC, DEN, V, AGG);

  // output projection straight into d_out
  gemm_bias_wmma<<<gGemm, blk, 0, stream>>>(AGG, Wo, bo, (float*)d_out, N_NODES);
}
